// EmbedMatcher_17635135717976
// MI455X (gfx1250) — compile-verified, matrix operand order checked by reading, not run on hardware
//
#include <hip/hip_runtime.h>
#include <hip/hip_bf16.h>

// ---------------------------------------------------------------------------
// EmbedMatcher fused kernel for MI455X (gfx1250, wave32, WMMA)
//
// Math simplifications (exact w.r.t. the reference):
//  * softmax over a single column == 1  =>  r == s_row (constant per support)
//    => fold s_row @ W_hh[:,256:].T into the bias (per-support constant)
//  * c/h_full columns 256..511 are dead => compute only 4*256 gate columns
//
// Hot loop: G[B,1024] = [q|h][B,512] @ Wc^T, Wc = [W_ih | W_hh[:,:256]] bf16,
// v_wmma_f32_16x16x32_bf16; q/h/c resident in LDS; weights streamed from L2.
// Each wave owns a unique 32-col H slice and both 16-row M tiles, so every
// B fragment is loaded once per block and feeds two WMMAs.
// q gather goes through GLOBAL_LOAD_ASYNC_TO_LDS_B128 (ASYNCcnt) when present.
// ---------------------------------------------------------------------------

typedef __attribute__((ext_vector_type(16))) __bf16 v16bf;
typedef __attribute__((ext_vector_type(8)))  __bf16 v8bf;
typedef __attribute__((ext_vector_type(8)))  float  v8f;
typedef __attribute__((ext_vector_type(4)))  int    v4i;

#define MB       32      // batch rows per block
#define DQ       256     // D = 2*EMBED_DIM
#define HH       512     // H
#define G4       2048    // 4*H
#define KDIM     512     // K = [q|h]
#define NBATCH   16384

#if defined(__has_builtin)
# if __has_builtin(__builtin_amdgcn_global_load_async_to_lds_b128) && \
     __has_builtin(__builtin_amdgcn_s_wait_asynccnt)
#   define HAVE_ASYNC_LDS 1
# endif
#endif

#if defined(HAVE_ASYNC_LDS)
typedef __attribute__((address_space(1))) v4i gv4i;   // global 16B vector
typedef __attribute__((address_space(3))) v4i lv4i;   // LDS    16B vector
#endif

// ---- fragment loader: 16 bf16 per lane, two 16B loads (works for LDS+global)
template <typename T>
__device__ __forceinline__ v16bf ldfrag(const T* base, int row, int stride, int k) {
  const T* p = base + (size_t)row * stride + k;
  v8bf lo = *(const v8bf*)(p);
  v8bf hi = *(const v8bf*)(p + 16);
  return __builtin_shufflevector(lo, hi, 0,1,2,3,4,5,6,7,8,9,10,11,12,13,14,15);
}

__device__ __forceinline__ float sigm_f(float x) {
  return 1.0f / (1.0f + __expf(-x));
}
__device__ __forceinline__ float tanh_f(float x) {
  float e = __expf(2.0f * x);
  return (e - 1.0f) / (e + 1.0f);
}

// ---------------------------------------------------------------------------
// Prep kernels
// ---------------------------------------------------------------------------
__global__ void prep_wc(const float* __restrict__ Wih,
                        const float* __restrict__ Whh,
                        __bf16* __restrict__ Wc) {
  int i = blockIdx.x * 256 + threadIdx.x;           // 2048*512
  if (i >= G4 * KDIM) return;
  int n = i >> 9, k = i & 511;
  float v = (k < DQ) ? Wih[n * DQ + k] : Whh[n * HH + (k - DQ)];
  Wc[i] = (__bf16)v;
}

__global__ void prep_sfeat(const int* __restrict__ support,
                           const float* __restrict__ emb,
                           float* __restrict__ sfeat) {
  int i = blockIdx.x * 256 + threadIdx.x;           // 4*256
  if (i >= 4 * DQ) return;
  int s = i >> 8, f = i & 255;
  int idx = support[s * 2 + (f >> 7)];
  sfeat[i] = emb[(size_t)idx * 128 + (f & 127)];
}

// cst0[n] = b_ih+b_hh ; cstS[s][n] = cst0[n] + s_row . W_hh[n, 256:512]
__global__ void prep_cst(const float* __restrict__ bih,
                         const float* __restrict__ bhh,
                         const float* __restrict__ Whh,
                         const float* __restrict__ sfeat,
                         float* __restrict__ cst0,
                         float* __restrict__ cstS) {
  int i = blockIdx.x * 256 + threadIdx.x;           // 4*2048
  if (i >= 4 * G4) return;
  int s = i >> 11, n = i & (G4 - 1);
  float base = bih[n] + bhh[n];
  if (s == 0) cst0[n] = base;
  float acc = base;
  const float* sf = sfeat + s * DQ;
  const float* wr = Whh + (size_t)n * HH + DQ;
  #pragma unroll 4
  for (int j = 0; j < DQ; ++j) acc = fmaf(sf[j], wr[j], acc);
  cstS[i] = acc;
}

// ---------------------------------------------------------------------------
// Main fused kernel: 512 blocks x 256 threads (8 wave32 / block)
// ---------------------------------------------------------------------------
__global__ __launch_bounds__(256) void matcher_main(
    const int*   __restrict__ query,
    const float* __restrict__ emb,
    const __bf16* __restrict__ Wc,     // [2048][512] bf16, row n = gate col
    const float* __restrict__ cst0,    // [2048]
    const float* __restrict__ cstS,    // [4][2048]
    const float* __restrict__ sfeat,   // [4][256]
    float*       __restrict__ out)     // [16384]
{
  // A operand: cols 0..255 = bf16(q); 256..511 / 512..767 = h ping/pong
  __shared__ __align__(16) __bf16 sA[MB][768];   // 48 KB
  __shared__ __align__(16) float  sQ[MB][DQ];    // 32 KB (fp32 q for residual)
  __shared__ __align__(16) float  sC[MB][DQ];    // 32 KB (live c columns only)
  __shared__ float sScore[MB];
  __shared__ float sMax[MB];

  const int tid     = threadIdx.x;
  const int lane    = tid & 31;
  const int wave    = tid >> 5;
  const int rowBase = blockIdx.x * MB;

  // ---- phase 1: gather q = [emb[q0] | emb[q1]] into sQ (f32) ----
  const int gm    = tid >> 3;            // 0..31 (batch row in tile)
  const int gpart = tid & 7;             // 8 x 32-col chunks
  const int gcol  = gpart * 32;
  {
    int idx = query[(rowBase + gm) * 2 + (gcol >> 7)];
    const float* src = emb + (size_t)idx * 128 + (gcol & 127);
#if defined(HAVE_ASYNC_LDS)
    // ASYNCcnt-tracked DMA into LDS, no VGPR round-trip
    #pragma unroll
    for (int i = 0; i < 8; ++i) {
      __builtin_amdgcn_global_load_async_to_lds_b128(
          (gv4i*)(src + i * 4),
          (lv4i*)(&sQ[gm][gcol + i * 4]),
          0, 0);
    }
    __builtin_amdgcn_s_wait_asynccnt(0);
#else
    #pragma unroll 8
    for (int i = 0; i < 32; ++i) sQ[gm][gcol + i] = src[i];
#endif
    if (tid < MB) sMax[tid] = -3.4e38f;
  }
  __syncthreads();

  // ---- phase 2: convert sQ -> sA (bf16 A-operand, K cols 0..255) ----
  {
    #pragma unroll 8
    for (int i = 0; i < 32; ++i) sA[gm][gcol + i] = (__bf16)sQ[gm][gcol + i];
  }
  __syncthreads();

  const int hc0  = wave * 2;                 // each wave: unique 2 N-tiles
  const int sel  = (lane < 16) ? 0 : 8;      // K sub-range per lane half
  const int ar0  = (lane & 15);              // A row, M-tile 0
  const int ar1  = 16 + (lane & 15);         // A row, M-tile 1

  for (int s = 0; s < 4; ++s) {
    // ---- per-support reset: zero step-0 h buffer, c state, score ----
    {
      #pragma unroll 8
      for (int i = 0; i < 32; ++i) {
        sA[gm][DQ + gcol + i] = (__bf16)0.0f;
        sC[gm][gcol + i] = 0.0f;
      }
      if (tid < MB) sScore[tid] = 0.0f;
    }
    __syncthreads();

    const float* csS = cstS + s * G4;
    const float* sf  = sfeat + s * DQ;

    for (int step = 0; step < 4; ++step) {
      const int   roff = (step & 1) ? 512 : 256;   // h read buffer
      const int   woff = (step & 1) ? 256 : 512;   // h write buffer
      const float* cc  = (step == 0) ? cst0 : csS; // folded bias (+ r term)

      #pragma unroll
      for (int hc = hc0; hc < hc0 + 2; ++hc) {
        const int n0 = hc * 16;
        const int j  = n0 + (lane & 15);           // live H column (0..255)

        // accumulators (both M tiles) seeded with per-column constant
        v8f ai0, af0, ag0, ao0, ai1, af1, ag1, ao1;
        {
          float ci = cc[j], cf = cc[512 + j], cg = cc[1024 + j], co = cc[1536 + j];
          #pragma unroll
          for (int v = 0; v < 8; ++v) {
            ai0[v] = ci; af0[v] = cf; ag0[v] = cg; ao0[v] = co;
            ai1[v] = ci; af1[v] = cf; ag1[v] = cg; ao1[v] = co;
          }
        }

        // hint next weight tiles toward WGP/L2
        __builtin_prefetch(Wc + (size_t)(j + 16) * KDIM, 0, 3);
        __builtin_prefetch(Wc + (size_t)(1536 + j + 16) * KDIM, 0, 3);

        #pragma unroll 4
        for (int kk = 0; kk < 16; ++kk) {
          const int kb = kk * 32;
          const int ka = (kb < DQ) ? kb : (kb - DQ + roff);
          v16bf a0 = ldfrag(&sA[0][0], ar0,       768,  ka + sel);  // LDS
          v16bf a1 = ldfrag(&sA[0][0], ar1,       768,  ka + sel);  // LDS
          v16bf b0 = ldfrag(Wc,        j,         KDIM, kb + sel);  // i gate
          v16bf b1 = ldfrag(Wc,        512  + j,  KDIM, kb + sel);  // f gate
          v16bf b2 = ldfrag(Wc,        1024 + j,  KDIM, kb + sel);  // g gate
          v16bf b3 = ldfrag(Wc,        1536 + j,  KDIM, kb + sel);  // o gate
          ai0 = __builtin_amdgcn_wmma_f32_16x16x32_bf16(false, a0, false, b0, (short)0, ai0, false, false);
          ai1 = __builtin_amdgcn_wmma_f32_16x16x32_bf16(false, a1, false, b0, (short)0, ai1, false, false);
          af0 = __builtin_amdgcn_wmma_f32_16x16x32_bf16(false, a0, false, b1, (short)0, af0, false, false);
          af1 = __builtin_amdgcn_wmma_f32_16x16x32_bf16(false, a1, false, b1, (short)0, af1, false, false);
          ag0 = __builtin_amdgcn_wmma_f32_16x16x32_bf16(false, a0, false, b2, (short)0, ag0, false, false);
          ag1 = __builtin_amdgcn_wmma_f32_16x16x32_bf16(false, a1, false, b2, (short)0, ag1, false, false);
          ao0 = __builtin_amdgcn_wmma_f32_16x16x32_bf16(false, a0, false, b3, (short)0, ao0, false, false);
          ao1 = __builtin_amdgcn_wmma_f32_16x16x32_bf16(false, a1, false, b3, (short)0, ao1, false, false);
        }

        // ---- LSTM cell elementwise + residual + score (both M tiles) ----
        const float sfj = sf[j];
        #pragma unroll
        for (int v = 0; v < 8; ++v) {
          int m0 = v + ((lane < 16) ? 0 : 8);      // M-tile 0 row
          float cOld = sC[m0][j];
          float cNew = sigm_f(af0[v]) * cOld + sigm_f(ai0[v]) * tanh_f(ag0[v]);
          sC[m0][j] = cNew;
          float h = sQ[m0][j] + sigm_f(ao0[v]) * tanh_f(cNew);
          sA[m0][woff + j] = (__bf16)h;
          if (step == 3) atomicAdd(&sScore[m0], h * sfj);
        }
        #pragma unroll
        for (int v = 0; v < 8; ++v) {
          int m1 = 16 + v + ((lane < 16) ? 0 : 8); // M-tile 1 row
          float cOld = sC[m1][j];
          float cNew = sigm_f(af1[v]) * cOld + sigm_f(ai1[v]) * tanh_f(ag1[v]);
          sC[m1][j] = cNew;
          float h = sQ[m1][j] + sigm_f(ao1[v]) * tanh_f(cNew);
          sA[m1][woff + j] = (__bf16)h;
          if (step == 3) atomicAdd(&sScore[m1], h * sfj);
        }
      }
      __syncthreads();
    }

    if (tid < MB) sMax[tid] = fmaxf(sMax[tid], sScore[tid]);
    __syncthreads();
  }

  if (tid < MB) out[rowBase + tid] = sMax[tid];
}

// ---------------------------------------------------------------------------
// Launch
// ---------------------------------------------------------------------------
extern "C" void kernel_launch(void* const* d_in, const int* in_sizes, int n_in,
                              void* d_out, int out_size, void* d_ws, size_t ws_size,
                              hipStream_t stream) {
  (void)in_sizes; (void)n_in; (void)out_size; (void)ws_size;

  const int*   query   = (const int*)d_in[0];
  const int*   support = (const int*)d_in[1];
  const float* emb     = (const float*)d_in[2];
  const float* Wih     = (const float*)d_in[3];
  const float* Whh     = (const float*)d_in[4];
  const float* bih     = (const float*)d_in[5];
  const float* bhh     = (const float*)d_in[6];
  float*       out     = (float*)d_out;

  char* ws = (char*)d_ws;
  __bf16* Wc    = (__bf16*)(ws + 0);          // 2048*512*2 = 2,097,152 B
  float*  cst0  = (float*) (ws + 2097152);    // 2048*4
  float*  cstS  = (float*) (ws + 2105344);    // 4*2048*4
  float*  sfeat = (float*) (ws + 2138112);    // 4*256*4

  prep_wc   <<<4096, 256, 0, stream>>>(Wih, Whh, Wc);
  prep_sfeat<<<   4, 256, 0, stream>>>(support, emb, sfeat);
  prep_cst  <<<  32, 256, 0, stream>>>(bih, bhh, Whh, sfeat, cst0, cstS);
  matcher_main<<<NBATCH / MB, 256, 0, stream>>>(query, emb, Wc, cst0, cstS, sfeat, out);
}
